// YATE_Attention_43791486550327
// MI455X (gfx1250) — compile-verified, matrix operand order checked by reading, not run on hardware
//
#include <hip/hip_runtime.h>
#include <hip/hip_bf16.h>
#include <hip/hip_fp16.h>

#define DD 256
#define HH 8
#define CC 32
#define NN 10000
#define EE 200000

typedef __attribute__((ext_vector_type(16))) _Float16 v16h;
typedef __attribute__((ext_vector_type(8)))  float    v8f;

// K index of the first element of VGPR-pair v for this lane-half (ISA 16-bit A/B layout)
__device__ __forceinline__ int kpair(int v, int hi) {
    int kbase = hi * 8;
    return (v < 4) ? (kbase + 2 * v) : (kbase + 16 + 2 * (v - 4));
}

__device__ __forceinline__ void atomicMaxFloat(float* addr, float value) {
    if (value >= 0.0f)
        atomicMax((int*)addr, __float_as_int(value));
    else
        atomicMin((unsigned int*)addr, (unsigned int)__float_as_int(value));
}

// Native f32 atomic add (global_atomic_add_f32) — avoids CAS-loop lowering.
__device__ __forceinline__ void atomAddF(float* addr, float value) {
    unsafeAtomicAdd(addr, value);
}

// ---------------------------------------------------------------------------
// Kernel 1: init accumulators (node out = 0, m = -inf, denom = 0)
// ---------------------------------------------------------------------------
__global__ void yate_init_kernel(float* __restrict__ nodeOut,
                                 float* __restrict__ m,
                                 float* __restrict__ denom) {
    int i = blockIdx.x * blockDim.x + threadIdx.x;
    if (i < NN * DD) nodeOut[i] = 0.0f;
    if (i < NN * HH) { m[i] = -__builtin_inff(); denom[i] = 0.0f; }
}

// ---------------------------------------------------------------------------
// Kernel 2: q = x @ Wq^T + bq   (WMMA, one 16x32 tile per wave: two
// independent accumulator chains hide the WMMA->WMMA RAW hazard slots)
// waves total = 625 Mtiles * 8 Npairs = 5000  -> 625 blocks * 8 waves
// ---------------------------------------------------------------------------
__global__ void yate_gemm_q_kernel(const float* __restrict__ x,
                                   const float* __restrict__ Wq,
                                   const float* __restrict__ bq,
                                   float* __restrict__ q) {
    int wave  = (blockIdx.x * blockDim.x + threadIdx.x) >> 5;
    int lane  = threadIdx.x & 31;
    int tileM = wave >> 3;        // 0..624
    int npair = wave & 7;         // 32-column slab
    int lr = lane & 15, hi = lane >> 4;

    int row   = tileM * 16 + lr;
    int ncol0 = npair * 32 + lr;
    int ncol1 = ncol0 + 16;
    const float* xrow  = x  + (size_t)row   * DD;
    const float* wrow0 = Wq + (size_t)ncol0 * DD;   // B[k][n] = Wq[n*DD + k]
    const float* wrow1 = Wq + (size_t)ncol1 * DD;

    float b0 = bq[ncol0], b1 = bq[ncol1];
    v8f acc0, acc1;
#pragma unroll
    for (int i = 0; i < 8; ++i) { acc0[i] = b0; acc1[i] = b1; }

#pragma unroll
    for (int kk = 0; kk < 8; ++kk) {
        int k0 = kk * 32;
        v16h a, bm0, bm1;
#pragma unroll
        for (int v = 0; v < 8; ++v) {
            int k = k0 + kpair(v, hi);
            float2 fa = *(const float2*)(xrow  + k);
            float2 f0 = *(const float2*)(wrow0 + k);
            float2 f1 = *(const float2*)(wrow1 + k);
            a  [2*v] = (_Float16)fa.x; a  [2*v+1] = (_Float16)fa.y;
            bm0[2*v] = (_Float16)f0.x; bm0[2*v+1] = (_Float16)f0.y;
            bm1[2*v] = (_Float16)f1.x; bm1[2*v+1] = (_Float16)f1.y;
        }
        acc0 = __builtin_amdgcn_wmma_f32_16x16x32_f16(false, a, false, bm0,
                                                      (short)0, acc0, false, false);
        acc1 = __builtin_amdgcn_wmma_f32_16x16x32_f16(false, a, false, bm1,
                                                      (short)0, acc1, false, false);
    }

#pragma unroll
    for (int r = 0; r < 8; ++r) {
        int mrow = tileM * 16 + r + 8 * hi;
        q[(size_t)mrow * DD + ncol0] = acc0[r];
        q[(size_t)mrow * DD + ncol1] = acc1[r];
    }
}

// ---------------------------------------------------------------------------
// Kernel 3: fused Z = x[src] * edge_feat ; k = Z@Wk^T+bk ; v = Z@Wv^T+bv ;
//           edge = Z@We^T+be.  A-fragments (full K=256) cached in registers,
//           reused across 4 N-tiles x 3 matrices (12 WMMAs per A load);
//           3 interleaved accumulator chains hide WMMA RAW hazards.
// waves total = 12500 Mtiles * 4 quads = 50000 -> 6250 blocks * 8 waves
// ---------------------------------------------------------------------------
__global__ void yate_gemm_edge_kernel(const float* __restrict__ x,
                                      const float* __restrict__ ef,
                                      const float* __restrict__ Wk, const float* __restrict__ bk,
                                      const float* __restrict__ Wv, const float* __restrict__ bv,
                                      const float* __restrict__ We, const float* __restrict__ be,
                                      const int*   __restrict__ srcIdx,
                                      _Float16* __restrict__ kout,
                                      _Float16* __restrict__ vout,
                                      float*    __restrict__ eout) {
    int wave  = (blockIdx.x * blockDim.x + threadIdx.x) >> 5;
    int lane  = threadIdx.x & 31;
    int tileM = wave >> 2;        // 0..12499
    int quad  = wave & 3;         // 64-column slab
    int lr = lane & 15, hi = lane >> 4;

    int erow = tileM * 16 + lr;
    int s = srcIdx[erow];
    const float* xrow = x  + (size_t)s    * DD;
    const float* frow = ef + (size_t)erow * DD;

    // Cache all 8 K-step A fragments of Z (f16)
    v16h afrag[8];
#pragma unroll
    for (int kk = 0; kk < 8; ++kk) {
        int k0 = kk * 32;
        v16h a;
#pragma unroll
        for (int v = 0; v < 8; ++v) {
            int k = k0 + kpair(v, hi);
            float2 fx = *(const float2*)(xrow + k);
            float2 fe = *(const float2*)(frow + k);
            a[2*v]   = (_Float16)(fx.x * fe.x);
            a[2*v+1] = (_Float16)(fx.y * fe.y);
        }
        afrag[kk] = a;
    }

    for (int nt = 0; nt < 4; ++nt) {
        int ncol = quad * 64 + nt * 16 + lr;
        const float* wkr = Wk + (size_t)ncol * DD;
        const float* wvr = Wv + (size_t)ncol * DD;
        const float* wer = We + (size_t)ncol * DD;
        float bbk = bk[ncol], bbv = bv[ncol], bbe = be[ncol];

        v8f ak, av, ae;
#pragma unroll
        for (int i = 0; i < 8; ++i) { ak[i] = bbk; av[i] = bbv; ae[i] = bbe; }

#pragma unroll
        for (int kk = 0; kk < 8; ++kk) {
            int k0 = kk * 32;
            v16h bkf, bvf, bef;
#pragma unroll
            for (int v = 0; v < 8; ++v) {
                int k = k0 + kpair(v, hi);
                float2 f1 = *(const float2*)(wkr + k);
                float2 f2 = *(const float2*)(wvr + k);
                float2 f3 = *(const float2*)(wer + k);
                bkf[2*v] = (_Float16)f1.x; bkf[2*v+1] = (_Float16)f1.y;
                bvf[2*v] = (_Float16)f2.x; bvf[2*v+1] = (_Float16)f2.y;
                bef[2*v] = (_Float16)f3.x; bef[2*v+1] = (_Float16)f3.y;
            }
            ak = __builtin_amdgcn_wmma_f32_16x16x32_f16(false, afrag[kk], false, bkf,
                                                        (short)0, ak, false, false);
            av = __builtin_amdgcn_wmma_f32_16x16x32_f16(false, afrag[kk], false, bvf,
                                                        (short)0, av, false, false);
            ae = __builtin_amdgcn_wmma_f32_16x16x32_f16(false, afrag[kk], false, bef,
                                                        (short)0, ae, false, false);
        }

#pragma unroll
        for (int r = 0; r < 8; ++r) {
            int mrow = tileM * 16 + r + 8 * hi;
            size_t o = (size_t)mrow * DD + ncol;
            kout[o] = (_Float16)ak[r];
            vout[o] = (_Float16)av[r];
            eout[o] = ae[r];
        }
    }
}

// ---------------------------------------------------------------------------
// Kernel 4: score[e,h] = dot(q[dst[e],h,:], k[e,h,:]) / sqrt(C); segment max
// ---------------------------------------------------------------------------
__global__ void yate_score_kernel(const float* __restrict__ q,
                                  const _Float16* __restrict__ kbuf,
                                  const int* __restrict__ dstIdx,
                                  float* __restrict__ score,
                                  float* __restrict__ m) {
    int i = blockIdx.x * blockDim.x + threadIdx.x;   // E*H = 1.6M exactly
    int e = i >> 3, h = i & 7;
    int dn = dstIdx[e];
    const float*    qrow = q    + (size_t)dn * DD + h * CC;
    const _Float16* krow = kbuf + (size_t)e  * DD + h * CC;
    float s = 0.0f;
#pragma unroll
    for (int c = 0; c < CC; ++c) s += qrow[c] * (float)krow[c];
    s *= 0.17677669529663687f;   // 1/sqrt(32)
    score[i] = s;
    atomicMaxFloat(&m[dn * HH + h], s);
}

// ---------------------------------------------------------------------------
// Kernel 5: ex = exp(score - m[dst]); denom += ex; out[dst] += ex * v
// ---------------------------------------------------------------------------
__global__ void yate_scatter_kernel(const float* __restrict__ score,
                                    const float* __restrict__ m,
                                    const _Float16* __restrict__ vbuf,
                                    const int* __restrict__ dstIdx,
                                    float* __restrict__ denom,
                                    float* __restrict__ nodeOut) {
    int i = blockIdx.x * blockDim.x + threadIdx.x;
    int e = i >> 3, h = i & 7;
    int dn = dstIdx[e];
    float ex = __expf(score[i] - m[dn * HH + h]);
    atomAddF(&denom[dn * HH + h], ex);
    const _Float16* vrow = vbuf + (size_t)e * DD + h * CC;
    float* orow = nodeOut + (size_t)dn * DD + h * CC;
#pragma unroll
    for (int c = 0; c < CC; ++c) atomAddF(&orow[c], ex * (float)vrow[c]);
}

// ---------------------------------------------------------------------------
// Kernel 6: out /= max(denom, 1e-16)
// ---------------------------------------------------------------------------
__global__ void yate_norm_kernel(float* __restrict__ nodeOut,
                                 const float* __restrict__ denom) {
    int i = blockIdx.x * blockDim.x + threadIdx.x;   // N*D exactly
    int n = i / DD;
    int h = (i % DD) / CC;
    nodeOut[i] = nodeOut[i] / fmaxf(denom[n * HH + h], 1e-16f);
}

// ---------------------------------------------------------------------------
extern "C" void kernel_launch(void* const* d_in, const int* in_sizes, int n_in,
                              void* d_out, int out_size, void* d_ws, size_t ws_size,
                              hipStream_t stream) {
    const float* x   = (const float*)d_in[0];
    const float* ef  = (const float*)d_in[1];
    const float* Wq  = (const float*)d_in[2];
    const float* bq  = (const float*)d_in[3];
    const float* Wk  = (const float*)d_in[4];
    const float* bk  = (const float*)d_in[5];
    const float* Wv  = (const float*)d_in[6];
    const float* bv  = (const float*)d_in[7];
    const float* We  = (const float*)d_in[8];
    const float* be  = (const float*)d_in[9];
    const int*   eidx = (const int*)d_in[10];
    const int* src = eidx;        // edge_index[0]
    const int* dst = eidx + EE;   // edge_index[1]

    float* nodeOut = (float*)d_out;                 // [N, D]
    float* edgeOut = nodeOut + (size_t)NN * DD;     // [E, D]

    // Workspace layout (bytes):
    //   q     f32 [N,D]   :       0 ..  10,240,000
    //   k     f16 [E,D]   :  10,240,000 .. 112,640,000
    //   v     f16 [E,D]   : 112,640,000 .. 215,040,000
    //   score f32 [E,H]   : 215,040,000 .. 221,440,000
    //   m     f32 [N,H]   : 221,440,000 .. 221,760,000
    //   denom f32 [N,H]   : 221,760,000 .. 222,080,000
    char* ws = (char*)d_ws;
    float*    q     = (float*)   (ws);
    _Float16* kbuf  = (_Float16*)(ws + (size_t)10240000);
    _Float16* vbuf  = (_Float16*)(ws + (size_t)112640000);
    float*    score = (float*)   (ws + (size_t)215040000);
    float*    m     = (float*)   (ws + (size_t)221440000);
    float*    denom = (float*)   (ws + (size_t)221760000);

    // 1. init accumulators (must re-run every call for determinism)
    yate_init_kernel<<<(NN * DD) / 256, 256, 0, stream>>>(nodeOut, m, denom);
    // 2. q projection (5000 wave-tasks, 2 N-tiles each / 8 waves per block)
    yate_gemm_q_kernel<<<625, 256, 0, stream>>>(x, Wq, bq, q);
    // 3. fused Z -> k, v, edge (50000 wave-tasks / 8 per block)
    yate_gemm_edge_kernel<<<6250, 256, 0, stream>>>(x, ef, Wk, bk, Wv, bv, We, be,
                                                    src, kbuf, vbuf, edgeOut);
    // 4. attention logits + segment max (E*H = 1.6M threads)
    yate_score_kernel<<<6250, 256, 0, stream>>>(q, kbuf, dst, score, m);
    // 5. exp + scatter (denom and weighted values)
    yate_scatter_kernel<<<6250, 256, 0, stream>>>(score, m, vbuf, dst, denom, nodeOut);
    // 6. normalize
    yate_norm_kernel<<<(NN * DD) / 256, 256, 0, stream>>>(nodeOut, denom);
}
